// SwinTransformerBlock_59407987638610
// MI455X (gfx1250) — matrix operand order repaired and checked by reading, not compiled, not run on hardware
//
#include <hip/hip_runtime.h>
#include <math.h>

// ---------------------------------------------------------------------------
// Types / WMMA helpers (CDNA5 gfx1250, wave32, v_wmma_f32_16x16x32_f16)
// ---------------------------------------------------------------------------
typedef __attribute__((ext_vector_type(16))) _Float16     v16h;
typedef __attribute__((ext_vector_type(8)))  float        v8f;
typedef __attribute__((ext_vector_type(4)))  unsigned int u32x4;
typedef __attribute__((ext_vector_type(4)))  float        f32x4;

union Frag16 { v16h v; _Float16 h[16]; u32x4 q[2]; };
union Acc8   { v8f  v; float f[8]; };

__device__ __forceinline__ v8f zero8() {
    v8f z = {0.f,0.f,0.f,0.f,0.f,0.f,0.f,0.f};
    return z;
}

// A-matrix fragment (16x32 f16). Lane L: row M = L&15, group g = L>>4.
// Element e -> K = k0 + 16*g + e (contiguous)
// `rowk0` points at &A[M][k0] of a row-major matrix.
__device__ __forceinline__ v16h afrag(const _Float16* rowk0, int g) {
    Frag16 f;
    f.q[0] = *(const u32x4*)(rowk0 + 16*g);
    f.q[1] = *(const u32x4*)(rowk0 + 16*g + 8);
    return f.v;
}

// B-matrix fragment (32x16 f16). Lane L: col N = L&15, group g = L>>4.
// Element e -> K = k0 + 16*g + e (contiguous). `p` points at 16 contiguous
// K-values for this lane's column, i.e. &WT[N][k0 + 16*g] of W^T row-major.
__device__ __forceinline__ v16h bfrag(const _Float16* p) {
    Frag16 f;
    f.q[0] = *(const u32x4*)(p);
    f.q[1] = *(const u32x4*)(p + 8);
    return f.v;
}

__device__ __forceinline__ v8f wmma16(v16h a, v16h b, v8f c) {
    return __builtin_amdgcn_wmma_f32_16x16x32_f16(false, a, false, b,
                                                  (short)0, c, false, false);
}

// ---------------------------------------------------------------------------
// Problem constants
// ---------------------------------------------------------------------------
#define C_DIM   128
#define HID     512
#define NHEAD   4
#define HDIM    32
#define WS7     7
#define NTOK    49      // tokens per window
#define HH_     56
#define BATCH   64
#define NWIN    64      // windows per image (8x8)
#define SCALE_QK 0.17677669529663687f   // 1/sqrt(32)

// f16 transposed-weight workspace layout (element offsets)
#define QKVT_OFF 0                  // [384][128]
#define PROJT_OFF 49152             // [128][128]
#define W1T_OFF   65536             // [512][128]
#define W2T_OFF   131072            // [128][512]
#define WT_TOTAL  196608

// ---------------------------------------------------------------------------
// Kernel 0: convert fp32 weights -> f16, transposed to [N][K] row-major
// ---------------------------------------------------------------------------
__global__ __launch_bounds__(256)
void swin_cvt_weights(const float* __restrict__ qkv_w,   // [128][384]
                      const float* __restrict__ proj_w,  // [128][128]
                      const float* __restrict__ w1,      // [128][512]
                      const float* __restrict__ w2,      // [512][128]
                      _Float16* __restrict__ wT) {
    int idx = blockIdx.x * 256 + threadIdx.x;
    if (idx >= WT_TOTAL) return;
    float v;
    if (idx < PROJT_OFF) {                 // qkvT [384][128]
        int n = idx >> 7, k = idx & 127;
        v = qkv_w[k * 384 + n];
    } else if (idx < W1T_OFF) {            // projT [128][128]
        int l = idx - PROJT_OFF;
        int n = l >> 7, k = l & 127;
        v = proj_w[k * 128 + n];
    } else if (idx < W2T_OFF) {            // w1T [512][128]
        int l = idx - W1T_OFF;
        int n = l >> 7, k = l & 127;
        v = w1[k * 512 + n];
    } else {                               // w2T [128][512]
        int l = idx - W2T_OFF;
        int n = l >> 9, k = l & 511;
        v = w2[k * 128 + n];
    }
    wT[idx] = (_Float16)v;
}

// ---------------------------------------------------------------------------
// Kernel 1: fused shifted-window attention. One block per window (4096),
// 256 threads = 8 wave32. Writes xres = x + attn_out into `out`.
// ---------------------------------------------------------------------------
__global__ __launch_bounds__(256)
void swin_attn(const float* __restrict__ x,        // [64][3136][128]
               const float* __restrict__ qkv_b,    // [384]
               const float* __restrict__ proj_b,   // [128]
               const float* __restrict__ rel_tab,  // [169][4]
               const _Float16* __restrict__ wT,
               float* __restrict__ out) {
    // LDS (~129 KB): window activations + per-head attention state
    __shared__ _Float16 sX [64 * 128];        // phase 0-1: X(f16); phase 4-5: O(f16)
    __shared__ _Float16 sQ [NHEAD * 64 * 32]; // [h][tok][dim]
    __shared__ _Float16 sK [NHEAD * 64 * 32]; // [h][tok][dim]
    __shared__ _Float16 sVT[NHEAD * 32 * 64]; // [h][dim][tok]  (transposed V)
    __shared__ _Float16 sS [NHEAD * 64 * 64]; // raw scores
    __shared__ _Float16 sP [NHEAD * 64 * 64]; // exp(scores-max), unnormalized
    __shared__ float    sRcp[NHEAD * 64];     // 1/rowsum
    __shared__ int      sCnt[64];             // shift-mask region id per token

    const int tid  = threadIdx.x;
    const int lane = tid & 31;
    const int wv   = tid >> 5;          // wave id 0..7
    const int g    = lane >> 4;         // lane group
    const int lm   = lane & 15;

    const int w   = blockIdx.x;
    const int b   = w >> 6;
    const int win = w & 63;
    const int wh  = win >> 3;
    const int ww  = win & 7;

    // ---- Phase 0: gather shifted window -> sX (f16), zero pad rows --------
    {
        int i  = tid >> 2;        // padded token row 0..63
        int c0 = (tid & 3) * 32;  // 32 channels per thread
        _Float16* dst = sX + i * 128 + c0;
        if (i < NTOK) {
            int ty = i / 7, tx = i % 7;
            int y  = wh * 7 + ty + 3; if (y  >= HH_) y  -= HH_;
            int xx = ww * 7 + tx + 3; if (xx >= HH_) xx -= HH_;
            const float* src = x + ((size_t)b * 3136 + y * 56 + xx) * 128 + c0;
            #pragma unroll
            for (int u = 0; u < 32; u += 8) {
                f32x4 f0 = *(const f32x4*)(src + u);
                f32x4 f1 = *(const f32x4*)(src + u + 4);
                Frag16 hh;
                hh.h[0]=(_Float16)f0.x; hh.h[1]=(_Float16)f0.y;
                hh.h[2]=(_Float16)f0.z; hh.h[3]=(_Float16)f0.w;
                hh.h[4]=(_Float16)f1.x; hh.h[5]=(_Float16)f1.y;
                hh.h[6]=(_Float16)f1.z; hh.h[7]=(_Float16)f1.w;
                *(u32x4*)(dst + u) = hh.q[0];
            }
        } else {
            u32x4 z = {0u,0u,0u,0u};
            #pragma unroll
            for (int u = 0; u < 32; u += 8) *(u32x4*)(dst + u) = z;
        }
        if (tid < NTOK) {   // shift-mask region id (regions split at 49/53)
            int ty = tid / 7, tx = tid % 7;
            int yy = wh * 7 + ty, xs = ww * 7 + tx;
            int gy = (yy < 49) ? 0 : ((yy < 53) ? 1 : 2);
            int gx = (xs < 49) ? 0 : ((xs < 53) ? 1 : 2);
            sCnt[tid] = gy * 3 + gx;
        }
    }
    __syncthreads();

    // ---- Phase 1: QKV GEMM  (64 x 384 x 128) ------------------------------
    {
        const _Float16* wq = wT + QKVT_OFF;     // [384][128]
        for (int nn = wv; nn < 24; nn += 8) {
            int ncol = nn * 16 + lm;            // 0..383 (this lane's column)
            v8f acc[4] = {zero8(), zero8(), zero8(), zero8()};
            #pragma unroll
            for (int ks = 0; ks < 4; ++ks) {
                v16h bf = bfrag(wq + ncol * 128 + ks * 32 + 16 * g);
                #pragma unroll
                for (int mt = 0; mt < 4; ++mt) {
                    v16h af = afrag(sX + (mt * 16 + lm) * 128 + ks * 32, g);
                    acc[mt] = wmma16(af, bf, acc[mt]);
                }
            }
            float bias = qkv_b[ncol];
            #pragma unroll
            for (int mt = 0; mt < 4; ++mt) {
                Acc8 a; a.v = acc[mt];
                int m0 = mt * 16 + 8 * g;       // rows m0..m0+7 (tokens)
                if (ncol < 256) {               // Q or K -> [h][tok][dim]
                    int head = (ncol >> 5) & 3;
                    int dim  = ncol & 31;
                    _Float16* dstb = (ncol < 128 ? sQ : sK) + (head * 64) * 32 + dim;
                    #pragma unroll
                    for (int e = 0; e < 8; ++e)
                        dstb[(m0 + e) * 32] = (_Float16)(a.f[e] + bias);
                } else {                        // V -> transposed [h][dim][tok]
                    int head = (ncol - 256) >> 5;
                    int dim  = ncol & 31;
                    Frag16 pk;
                    #pragma unroll
                    for (int e = 0; e < 8; ++e) pk.h[e] = (_Float16)(a.f[e] + bias);
                    *(u32x4*)(sVT + (head * 32 + dim) * 64 + m0) = pk.q[0];
                }
            }
        }
    }
    __syncthreads();

    // ---- Phase 2: S = Q*K^T per head (2 waves/head, 2 row-tiles each) -----
    {
        int head = wv >> 1;
        int mlo  = (wv & 1) * 2;
        const _Float16* Qh = sQ + head * 64 * 32;
        const _Float16* Kh = sK + head * 64 * 32;
        _Float16* Sh = sS + head * 64 * 64;
        #pragma unroll
        for (int nt = 0; nt < 4; ++nt) {
            // B[kdim][ntok] = K[ntok][kdim]: contiguous in kdim -> row of sK
            v16h bf = bfrag(Kh + (nt * 16 + lm) * 32 + 16 * g);
            #pragma unroll
            for (int mi = 0; mi < 2; ++mi) {
                int mt = mlo + mi;
                v16h af = afrag(Qh + (mt * 16 + lm) * 32, g);
                v8f acc = zero8();
                acc = wmma16(af, bf, acc);
                Acc8 s; s.v = acc;
                _Float16* dst = Sh + (mt * 16 + 8 * g) * 64 + nt * 16 + lm;
                #pragma unroll
                for (int e = 0; e < 8; ++e) dst[e * 64] = (_Float16)s.f[e];
            }
        }
    }
    __syncthreads();

    // ---- Phase 3: softmax rows (scale + rel-pos bias + shift mask) --------
    {
        int head = tid >> 6;
        int i    = tid & 63;
        if (i < NTOK) {
            const _Float16* srow = sS + head * 4096 + i * 64;
            _Float16*       prow = sP + head * 4096 + i * 64;
            int ty = i / 7, tx = i % 7;
            int cnt_i = sCnt[i];
            float mx = -1e30f;
            int jy = 0, jx = 0;
            for (int j = 0; j < NTOK; ++j) {
                float s = (float)srow[j] * SCALE_QK;
                int idx = (ty - jy + 6) * 13 + (tx - jx + 6);
                s += rel_tab[idx * 4 + head];
                if (cnt_i != sCnt[j]) s -= 100.f;
                prow[j] = (_Float16)s;
                mx = fmaxf(mx, s);
                if (++jx == 7) { jx = 0; ++jy; }
            }
            float sum = 0.f;
            for (int j = 0; j < NTOK; ++j) {
                float p = __expf((float)prow[j] - mx);
                sum += p;
                prow[j] = (_Float16)p;
            }
            for (int j = NTOK; j < 64; ++j) prow[j] = (_Float16)0.f;
            sRcp[head * 64 + i] = 1.f / sum;
        }
    }
    __syncthreads();

    // ---- Phase 4: O = (P/rowsum) * V  -> sX reused as O [64][128] ---------
    {
        int head = wv >> 1;
        int mlo  = (wv & 1) * 2;
        const _Float16* Ph = sP + head * 4096;        // [64][64]
        const _Float16* Vh = sVT + head * 32 * 64;    // [dim][tok]
        const float*    Rp = sRcp + head * 64;
        #pragma unroll
        for (int nt = 0; nt < 2; ++nt) {
            int dim = nt * 16 + lm;
            #pragma unroll
            for (int mi = 0; mi < 2; ++mi) {
                int mt = mlo + mi;
                v8f acc = zero8();
                #pragma unroll
                for (int ks = 0; ks < 2; ++ks) {
                    v16h af = afrag(Ph + (mt * 16 + lm) * 64 + ks * 32, g);
                    v16h bf = bfrag(Vh + dim * 64 + ks * 32 + 16 * g);
                    acc = wmma16(af, bf, acc);
                }
                Acc8 a; a.v = acc;
                int m0 = mt * 16 + 8 * g;
                _Float16* dst = sX + m0 * 128 + head * 32 + dim;
                #pragma unroll
                for (int e = 0; e < 8; ++e)
                    dst[e * 128] = (_Float16)(a.f[e] * Rp[m0 + e]);
            }
        }
    }
    __syncthreads();

    // ---- Phase 5: proj GEMM + residual + inverse-shift scatter ------------
    {
        const _Float16* wp = wT + PROJT_OFF;    // [128][128]
        int ncol = wv * 16 + lm;                // 0..127
        v8f acc[4] = {zero8(), zero8(), zero8(), zero8()};
        #pragma unroll
        for (int ks = 0; ks < 4; ++ks) {
            v16h bf = bfrag(wp + ncol * 128 + ks * 32 + 16 * g);
            #pragma unroll
            for (int mt = 0; mt < 4; ++mt) {
                v16h af = afrag(sX + (mt * 16 + lm) * 128 + ks * 32, g);
                acc[mt] = wmma16(af, bf, acc[mt]);
            }
        }
        float pb = proj_b[ncol];
        #pragma unroll
        for (int mt = 0; mt < 4; ++mt) {
            Acc8 a; a.v = acc[mt];
            int m0 = mt * 16 + 8 * g;
            #pragma unroll
            for (int e = 0; e < 8; ++e) {
                int i = m0 + e;
                if (i < NTOK) {
                    int ty = i / 7, tx = i % 7;
                    int y  = wh * 7 + ty + 3; if (y  >= HH_) y  -= HH_;
                    int xx = ww * 7 + tx + 3; if (xx >= HH_) xx -= HH_;
                    size_t off = ((size_t)b * 3136 + y * 56 + xx) * 128 + ncol;
                    out[off] = x[off] + a.f[e] + pb;   // xres
                }
            }
        }
    }
}

// ---------------------------------------------------------------------------
// Kernel 2: fused MLP: h = GELU(LN(xres)W1+b1)W2+b2; out = xres + h (inplace)
// One block per 64 rows (3136 blocks), 256 threads = 8 waves.
// ---------------------------------------------------------------------------
__global__ __launch_bounds__(256)
void swin_mlp(const float* __restrict__ n2g, const float* __restrict__ n2b,
              const float* __restrict__ b1,  const float* __restrict__ b2,
              const _Float16* __restrict__ wT,
              float* __restrict__ io) {
    __shared__ float    sXf[64 * 128];   // xres rows (residual source)
    __shared__ _Float16 sXn[64 * 128];   // layernormed f16
    __shared__ _Float16 sH [64 * 512];   // hidden f16
    __shared__ float    sMu[64], sRs[64];

    const int tid  = threadIdx.x;
    const int lane = tid & 31;
    const int wv   = tid >> 5;
    const int g    = lane >> 4;
    const int lm   = lane & 15;
    const size_t row0 = (size_t)blockIdx.x * 64;
    const float* src = io + row0 * 128;

    // load 64x128 fp32
    #pragma unroll
    for (int k = 0; k < 8; ++k) {
        int f4 = tid + 256 * k;
        *(f32x4*)(sXf + f4 * 4) = *(const f32x4*)(src + f4 * 4);
    }
    __syncthreads();

    // layernorm stats
    if (tid < 64) {
        const float* r = sXf + tid * 128;
        float s = 0.f, ss = 0.f;
        for (int c = 0; c < 128; ++c) { float v = r[c]; s += v; ss += v * v; }
        float mu = s * (1.f / 128.f);
        float var = ss * (1.f / 128.f) - mu * mu;
        sMu[tid] = mu;
        sRs[tid] = rsqrtf(var + 1e-5f);
    }
    __syncthreads();

    // normalize -> f16
    #pragma unroll
    for (int k = 0; k < 32; ++k) {
        int idx = tid + 256 * k;
        int r = idx >> 7, c = idx & 127;
        float xn = (sXf[idx] - sMu[r]) * sRs[r] * n2g[c] + n2b[c];
        sXn[idx] = (_Float16)xn;
    }
    __syncthreads();

    // GEMM1 (64 x 512 x 128) + exact GELU
    {
        const _Float16* w1T = wT + W1T_OFF;   // [512][128]
        for (int nn = wv; nn < 32; nn += 8) {
            int ncol = nn * 16 + lm;
            v8f acc[4] = {zero8(), zero8(), zero8(), zero8()};
            #pragma unroll
            for (int ks = 0; ks < 4; ++ks) {
                v16h bf = bfrag(w1T + ncol * 128 + ks * 32 + 16 * g);
                #pragma unroll
                for (int mt = 0; mt < 4; ++mt) {
                    v16h af = afrag(sXn + (mt * 16 + lm) * 128 + ks * 32, g);
                    acc[mt] = wmma16(af, bf, acc[mt]);
                }
            }
            float bb = b1[ncol];
            #pragma unroll
            for (int mt = 0; mt < 4; ++mt) {
                Acc8 a; a.v = acc[mt];
                int m0 = mt * 16 + 8 * g;
                #pragma unroll
                for (int e = 0; e < 8; ++e) {
                    float v = a.f[e] + bb;
                    v = 0.5f * v * (1.f + erff(v * 0.70710678118654752f));
                    sH[(m0 + e) * 512 + ncol] = (_Float16)v;
                }
            }
        }
    }
    __syncthreads();

    // GEMM2 (64 x 128 x 512) + residual, write back in place
    {
        const _Float16* w2T = wT + W2T_OFF;   // [128][512]
        int ncol = wv * 16 + lm;              // 0..127
        v8f acc[4] = {zero8(), zero8(), zero8(), zero8()};
        #pragma unroll
        for (int ks = 0; ks < 16; ++ks) {
            v16h bf = bfrag(w2T + ncol * 512 + ks * 32 + 16 * g);
            #pragma unroll
            for (int mt = 0; mt < 4; ++mt) {
                v16h af = afrag(sH + (mt * 16 + lm) * 512 + ks * 32, g);
                acc[mt] = wmma16(af, bf, acc[mt]);
            }
        }
        float bb = b2[ncol];
        #pragma unroll
        for (int mt = 0; mt < 4; ++mt) {
            Acc8 a; a.v = acc[mt];
            int m0 = mt * 16 + 8 * g;
            #pragma unroll
            for (int e = 0; e < 8; ++e) {
                int r = m0 + e;
                io[(row0 + r) * 128 + ncol] = sXf[r * 128 + ncol] + a.f[e] + bb;
            }
        }
    }
}

// ---------------------------------------------------------------------------
// Launch
// ---------------------------------------------------------------------------
extern "C" void kernel_launch(void* const* d_in, const int* in_sizes, int n_in,
                              void* d_out, int out_size, void* d_ws, size_t ws_size,
                              hipStream_t stream) {
    (void)in_sizes; (void)n_in; (void)out_size; (void)ws_size;
    const float* x       = (const float*)d_in[0];
    const float* qkv_w   = (const float*)d_in[1];
    const float* qkv_b   = (const float*)d_in[2];
    const float* proj_w  = (const float*)d_in[3];
    const float* proj_b  = (const float*)d_in[4];
    const float* rel_tab = (const float*)d_in[5];
    const float* n2g     = (const float*)d_in[6];
    const float* n2b     = (const float*)d_in[7];
    const float* w1      = (const float*)d_in[8];
    const float* b1      = (const float*)d_in[9];
    const float* w2      = (const float*)d_in[10];
    const float* b2      = (const float*)d_in[11];
    float* out = (float*)d_out;
    _Float16* wT = (_Float16*)d_ws;   // 196608 halves = 384 KB

    swin_cvt_weights<<<dim3((WT_TOTAL + 255) / 256), dim3(256), 0, stream>>>(
        qkv_w, proj_w, w1, w2, wT);
    swin_attn<<<dim3(4096), dim3(256), 0, stream>>>(
        x, qkv_b, proj_b, rel_tab, wT, out);
    swin_mlp<<<dim3(3136), dim3(256), 0, stream>>>(
        n2g, n2b, b1, b2, wT, out);
}
